// StructuredPositionPointerEncoder_7730941132934
// MI455X (gfx1250) — matrix operand order repaired and checked by reading, not compiled, not run on hardware
//
#include <hip/hip_runtime.h>
#include <math.h>

// Problem constants (from reference): H=1024, N=2048, B=64, P=4, T=16, heads=4, hd=256
#define HD   1024
#define NTOK 2048
#define NB   64
#define NP   4

typedef __attribute__((ext_vector_type(2)))  float        v2f;
typedef __attribute__((ext_vector_type(8)))  float        v8f;
typedef __attribute__((ext_vector_type(16))) _Float16     v16h;
typedef __attribute__((ext_vector_type(4)))  unsigned int u32x4;
typedef __attribute__((ext_vector_type(8)))  int          i32x8;
typedef __attribute__((ext_vector_type(4)))  int          i32x4;

#if defined(__HIP_DEVICE_COMPILE__) && __has_builtin(__builtin_amdgcn_wmma_f32_16x16x4_f32)
#define WMMA_F32 1
#else
#define WMMA_F32 0
#endif

#if defined(__HIP_DEVICE_COMPILE__) && __has_builtin(__builtin_amdgcn_tensor_load_to_lds)
#define HAVE_TDM 1
#else
#define HAVE_TDM 0
#endif

__device__ __forceinline__ float gelu_exact(float x) {
  return 0.5f * x * (1.0f + erff(x * 0.7071067811865475f));
}

__device__ __forceinline__ float block_sum(float v, float* s, int tid) {
  s[tid] = v; __syncthreads();
  for (int off = 128; off > 0; off >>= 1) {
    if (tid < off) s[tid] += s[tid + off];
    __syncthreads();
  }
  float r = s[0]; __syncthreads();
  return r;
}
__device__ __forceinline__ float block_max(float v, float* s, int tid) {
  s[tid] = v; __syncthreads();
  for (int off = 128; off > 0; off >>= 1) {
    if (tid < off) s[tid] = fmaxf(s[tid], s[tid + off]);
    __syncthreads();
  }
  float r = s[0]; __syncthreads();
  return r;
}

// ---------------------------------------------------------------------------
// Fused pointer stage: logits -> mask -> softmax over N -> attn @ spa -> LN.
// One block per batch; spa_feat[b] (8MB) second pass hits L2.
// ---------------------------------------------------------------------------
__global__ __launch_bounds__(256)
void pointer_stage(const float* __restrict__ spa, const int* __restrict__ mask,
                   const float* __restrict__ ps_w, const float* __restrict__ ps_b,
                   const float* __restrict__ pn_g, const float* __restrict__ pn_b,
                   float* __restrict__ out_pt /* (B,P,H) */) {
  __shared__ float s_w[NP * HD];      // 16 KB  (ps_w)
  __shared__ float s_att[NP * NTOK];  // 32 KB  (logits -> attn weights)
  __shared__ float s_red[256];

  const int b = blockIdx.x, tid = threadIdx.x;
  const float* sb = spa + (size_t)b * NTOK * HD;

  for (int i = tid; i < NP * HD; i += 256) s_w[i] = ps_w[i];
  __syncthreads();

  const float pb0 = ps_b[0], pb1 = ps_b[1], pb2 = ps_b[2], pb3 = ps_b[3];

  float m0 = -3.0e38f, m1 = -3.0e38f, m2 = -3.0e38f, m3 = -3.0e38f;
  for (int ni = 0; ni < NTOK / 256; ++ni) {
    const int n = ni * 256 + tid;
    const float* x = sb + (size_t)n * HD;
    float a0 = pb0, a1 = pb1, a2 = pb2, a3 = pb3;
    for (int k = 0; k < HD; k += 4) {
      const float4 xv = *(const float4*)(x + k);
      const float* w0 = &s_w[k];
      const float* w1 = &s_w[HD + k];
      const float* w2 = &s_w[2 * HD + k];
      const float* w3 = &s_w[3 * HD + k];
      a0 += xv.x * w0[0] + xv.y * w0[1] + xv.z * w0[2] + xv.w * w0[3];
      a1 += xv.x * w1[0] + xv.y * w1[1] + xv.z * w1[2] + xv.w * w1[3];
      a2 += xv.x * w2[0] + xv.y * w2[1] + xv.z * w2[2] + xv.w * w2[3];
      a3 += xv.x * w3[0] + xv.y * w3[1] + xv.z * w3[2] + xv.w * w3[3];
    }
    const bool mk = (mask[b * NTOK + n] != 0);
    const float l0 = mk ? a0 : -10000.0f;
    const float l1 = mk ? a1 : -10000.0f;
    const float l2 = mk ? a2 : -10000.0f;
    const float l3 = mk ? a3 : -10000.0f;
    s_att[n] = l0; s_att[NTOK + n] = l1; s_att[2 * NTOK + n] = l2; s_att[3 * NTOK + n] = l3;
    m0 = fmaxf(m0, l0); m1 = fmaxf(m1, l1); m2 = fmaxf(m2, l2); m3 = fmaxf(m3, l3);
  }
  __syncthreads();

  const float gm0 = block_max(m0, s_red, tid);
  const float gm1 = block_max(m1, s_red, tid);
  const float gm2 = block_max(m2, s_red, tid);
  const float gm3 = block_max(m3, s_red, tid);

  float s0 = 0.f, s1 = 0.f, s2 = 0.f, s3 = 0.f;
  for (int ni = 0; ni < NTOK / 256; ++ni) {
    const int n = ni * 256 + tid;
    float e;
    e = expf(s_att[n]            - gm0); s_att[n]            = e; s0 += e;
    e = expf(s_att[NTOK + n]     - gm1); s_att[NTOK + n]     = e; s1 += e;
    e = expf(s_att[2 * NTOK + n] - gm2); s_att[2 * NTOK + n] = e; s2 += e;
    e = expf(s_att[3 * NTOK + n] - gm3); s_att[3 * NTOK + n] = e; s3 += e;
  }
  const float inv0 = 1.0f / block_sum(s0, s_red, tid);
  const float inv1 = 1.0f / block_sum(s1, s_red, tid);
  const float inv2 = 1.0f / block_sum(s2, s_red, tid);
  const float inv3 = 1.0f / block_sum(s3, s_red, tid);
  for (int ni = 0; ni < NTOK / 256; ++ni) {
    const int n = ni * 256 + tid;
    s_att[n] *= inv0; s_att[NTOK + n] *= inv1;
    s_att[2 * NTOK + n] *= inv2; s_att[3 * NTOK + n] *= inv3;
  }
  __syncthreads();

  const int h0 = tid * 4;
  float4 acc[NP];
  for (int p = 0; p < NP; ++p) acc[p] = make_float4(0.f, 0.f, 0.f, 0.f);
  for (int n = 0; n < NTOK; ++n) {
    const float4 xv = *(const float4*)(sb + (size_t)n * HD + h0);
    const float w0 = s_att[n], w1 = s_att[NTOK + n];
    const float w2 = s_att[2 * NTOK + n], w3 = s_att[3 * NTOK + n];
    acc[0].x += w0 * xv.x; acc[0].y += w0 * xv.y; acc[0].z += w0 * xv.z; acc[0].w += w0 * xv.w;
    acc[1].x += w1 * xv.x; acc[1].y += w1 * xv.y; acc[1].z += w1 * xv.z; acc[1].w += w1 * xv.w;
    acc[2].x += w2 * xv.x; acc[2].y += w2 * xv.y; acc[2].z += w2 * xv.z; acc[2].w += w2 * xv.w;
    acc[3].x += w3 * xv.x; acc[3].y += w3 * xv.y; acc[3].z += w3 * xv.z; acc[3].w += w3 * xv.w;
  }

  float* outp = out_pt + (size_t)b * NP * HD;
  for (int p = 0; p < NP; ++p) {
    const float4 a = acc[p];
    const float sum = block_sum(a.x + a.y + a.z + a.w, s_red, tid);
    const float mu = sum * (1.0f / (float)HD);
    const float dx = a.x - mu, dy = a.y - mu, dz = a.z - mu, dw = a.w - mu;
    const float var = block_sum(dx * dx + dy * dy + dz * dz + dw * dw, s_red, tid);
    const float inv = rsqrtf(var * (1.0f / (float)HD) + 1e-5f);
    float* o = outp + p * HD + h0;
    o[0] = dx * inv * pn_g[h0 + 0] + pn_b[h0 + 0];
    o[1] = dy * inv * pn_g[h0 + 1] + pn_b[h0 + 1];
    o[2] = dz * inv * pn_g[h0 + 2] + pn_b[h0 + 2];
    o[3] = dw * inv * pn_g[h0 + 3] + pn_b[h0 + 3];
  }
}

// ---------------------------------------------------------------------------
// Generic WMMA GEMM:  C[M,N] = act( A[M,K4+Ktail] * B + bias )
//   transB=1: B[k][j] = Bw[j*ldb + k];  transB=0: B[k][j] = Bw[k*ldb + j]
// One wave per 16x16 tile; fp32 WMMA 16x16x4. Columns >= N are *clamped* on
// load (their D elements are never stored), so the K-loop has no divergence.
// M is a multiple of 16 at every call site.
// ---------------------------------------------------------------------------
__global__ __launch_bounds__(256)
void gemm_wmma(const float* __restrict__ A, int lda,
               const float* __restrict__ Bw, int ldb, int transB,
               const float* __restrict__ bias,
               float* __restrict__ C, int ldc,
               int M, int N, int K4, int Ktail, int fuseGelu) {
  const int wave = (int)((blockIdx.x * blockDim.x + threadIdx.x) >> 5);
  const int lane = threadIdx.x & 31;
  const int tilesN = (N + 15) >> 4;
  const int tilesM = (M + 15) >> 4;
  if (wave >= tilesM * tilesN) return;
  const int tm = wave / tilesN, tn = wave % tilesN;
  const int half = lane >> 4, lc = lane & 15;
  const int rowA = tm * 16 + lc;
  const int colB = tn * 16 + lc;
  const int colC = (colB < N) ? colB : (N - 1);  // clamp: no divergence in loop

  const float* ap = A + (size_t)rowA * lda;
  v8f c = {};

#if WMMA_F32
  if (transB) {
    const float* bp = Bw + (size_t)colC * ldb;
    #pragma unroll 4
    for (int kb = 0; kb < K4; kb += 4) {
      const int k0 = kb + half * 2;
      v2f a, b;
      a[0] = ap[k0]; a[1] = ap[k0 + 1];
      b[0] = bp[k0]; b[1] = bp[k0 + 1];
      c = __builtin_amdgcn_wmma_f32_16x16x4_f32(false, a, false, b, (short)0, c, false, false);
    }
  } else {
    const float* bp = Bw + colC;
    #pragma unroll 4
    for (int kb = 0; kb < K4; kb += 4) {
      const int k0 = kb + half * 2;
      v2f a, b;
      a[0] = ap[k0]; a[1] = ap[k0 + 1];
      b[0] = bp[(size_t)k0 * ldb]; b[1] = bp[(size_t)(k0 + 1) * ldb];
      c = __builtin_amdgcn_wmma_f32_16x16x4_f32(false, a, false, b, (short)0, c, false, false);
    }
  }
#else
  // f16 fallback (codegen-confirmed builtin): convert-on-load, f32 accumulate.
  for (int kb = 0; kb < K4; kb += 32) {
    v16h a, b;
    #pragma unroll
    for (int j = 0; j < 8; ++j) {                  // A 16x32 dense layout (ISA 7.12.2)
      const int kA = kb + half * 8 + j;
      const int kA2 = kA + 16;
      a[j]     = (_Float16)((kA  < K4) ? ap[kA ] : 0.f);
      a[8 + j] = (_Float16)((kA2 < K4) ? ap[kA2] : 0.f);
    }
    #pragma unroll
    for (int j = 0; j < 16; ++j) {                 // B 32x16: same K order as A
      const int kB = kb + half * 8 + (j & 7) + ((j >> 3) << 4);
      float bv = 0.f;
      if (kB < K4)
        bv = transB ? Bw[(size_t)colC * ldb + kB] : Bw[(size_t)kB * ldb + colC];
      b[j] = (_Float16)bv;
    }
    c = __builtin_amdgcn_wmma_f32_16x16x32_f16(false, a, false, b, (short)0, c, false, false);
  }
#endif

  // Ragged-K tail (e.g. the H+3 concat in the rtp projection)
  for (int kt = 0; kt < Ktail; ++kt) {
    const int k = K4 + kt;
    const float bv = transB ? Bw[(size_t)colC * ldb + k] : Bw[(size_t)k * ldb + colC];
    #pragma unroll
    for (int i = 0; i < 8; ++i) {
      const int m = tm * 16 + (half ? i + 8 : i);
      c[i] += A[(size_t)m * lda + k] * bv;
    }
  }

  // Epilogue: bias (+gelu); store per ISA C/D layout (VGPR i -> M=i or i+8)
  #pragma unroll
  for (int i = 0; i < 8; ++i) {
    const int m = tm * 16 + (half ? i + 8 : i);
    if (colB < N) {
      float v = c[i] + (bias ? bias[colB] : 0.0f);
      if (fuseGelu) v = gelu_exact(v);
      C[(size_t)m * ldc + colB] = v;
    }
  }
}

// ---------------------------------------------------------------------------
// Tiny 4-token multi-head attention (per batch). The whole 4x3072 qkv tile
// (48KB) is DMA'd to LDS with the Tensor Data Mover, then q.k / softmax /
// attn.v run out of LDS.  qkv: (B, 4, 3072) with [q|k|v] split.
// ---------------------------------------------------------------------------
__global__ __launch_bounds__(256)
void attn_stage(const float* __restrict__ qkv, float* __restrict__ ctx) {
  __shared__ float s_qkv[4 * 3072];  // 48 KB: [p][q|k|v]
  __shared__ float s_attn[4][4][4];  // [head][i][j]
  __shared__ float s_sc[64];
  const int b = blockIdx.x, tid = threadIdx.x;
  const float* gbase = qkv + (size_t)b * 4 * 3072;

#if HAVE_TDM
  if (tid < 32) {  // wave 0 issues the DMA (TDM ignores EXEC; per-wave instr)
    const unsigned long long ga = (unsigned long long)(uintptr_t)gbase;
    const unsigned lds = (unsigned)(unsigned long long)(uintptr_t)(&s_qkv[0]);
    u32x4 g0;
    g0[0] = 1u;                                           // count=1 (valid D#)
    g0[1] = lds;                                          // lds_addr (bytes)
    g0[2] = (unsigned)ga;                                 // global_addr[31:0]
    g0[3] = (unsigned)((ga >> 32) & 0x01FFFFFFu) | (2u << 30);  // addr[56:32] | type=2
    i32x8 g1;
    g1[0] = (int)(2u << 16);       // workgroup_mask=0, data_size=2 (4B)
    g1[1] = (int)(3072u << 16);    // tensor_dim0[15:0]=3072 in bits[31:16]
    g1[2] = (int)(4u << 16);       // tensor_dim0 hi=0; tensor_dim1=4
    g1[3] = (int)(3072u << 16);    // tensor_dim1 hi=0; tile_dim0=3072
    g1[4] = 4;                     // tile_dim1=4, tile_dim2=0
    g1[5] = 3072;                  // tensor_dim0_stride[31:0] = 3072
    g1[6] = 0; g1[7] = 0;          // stride hi / dim1_stride = 0
    i32x4 g2 = {0, 0, 0, 0}, g3 = {0, 0, 0, 0};
    i32x8 g4 = {0, 0, 0, 0, 0, 0, 0, 0};   // 6-arg toolchain variant: extra group
    __builtin_amdgcn_tensor_load_to_lds(g0, g1, g2, g3, g4, 0);
    __builtin_amdgcn_s_wait_tensorcnt(0);
  }
  __syncthreads();
#else
  for (int i = tid * 4; i < 4 * 3072; i += 256 * 4)
    *(float4*)(&s_qkv[i]) = *(const float4*)(gbase + i);
  __syncthreads();
#endif

  if (tid < 64) {
    const int h = tid >> 4, i = (tid >> 2) & 3, j = tid & 3;
    const float* q = &s_qkv[i * 3072 + h * 256];
    const float* k = &s_qkv[j * 3072 + 1024 + h * 256];
    float s = 0.f;
    for (int d = 0; d < 256; ++d) s += q[d] * k[d];
    s_sc[tid] = s * (1.0f / 16.0f);  // / sqrt(256)
  }
  __syncthreads();
  if (tid < 16) {
    const int h = tid >> 2, i = tid & 3;
    const float* sc = &s_sc[h * 16 + i * 4];
    float m = fmaxf(fmaxf(sc[0], sc[1]), fmaxf(sc[2], sc[3]));
    float e0 = expf(sc[0] - m), e1 = expf(sc[1] - m), e2 = expf(sc[2] - m), e3 = expf(sc[3] - m);
    const float inv = 1.0f / (e0 + e1 + e2 + e3);
    s_attn[h][i][0] = e0 * inv; s_attn[h][i][1] = e1 * inv;
    s_attn[h][i][2] = e2 * inv; s_attn[h][i][3] = e3 * inv;
  }
  __syncthreads();
  for (int i = 0; i < 4; ++i) {
    for (int d = tid; d < 1024; d += 256) {
      const int h = d >> 8;
      float s = 0.f;
      #pragma unroll
      for (int j = 0; j < 4; ++j) s += s_attn[h][i][j] * s_qkv[j * 3072 + 2048 + d];
      ctx[(size_t)b * 4096 + i * 1024 + d] = s;
    }
  }
}

// LayerNorm over H=1024 per row, optional residual: Y = LN(X [+ res]) * g + b
__global__ __launch_bounds__(256)
void ln1024(const float* __restrict__ X, const float* __restrict__ res,
            const float* __restrict__ g, const float* __restrict__ bt,
            float* __restrict__ Y) {
  __shared__ float s_red[256];
  const int row = blockIdx.x, tid = threadIdx.x;
  float4 v = *(const float4*)(X + (size_t)row * HD + tid * 4);
  if (res) {
    const float4 r = *(const float4*)(res + (size_t)row * HD + tid * 4);
    v.x += r.x; v.y += r.y; v.z += r.z; v.w += r.w;
  }
  const float sum = block_sum(v.x + v.y + v.z + v.w, s_red, tid);
  const float mu = sum * (1.0f / (float)HD);
  const float dx = v.x - mu, dy = v.y - mu, dz = v.z - mu, dw = v.w - mu;
  const float var = block_sum(dx * dx + dy * dy + dz * dz + dw * dw, s_red, tid);
  const float inv = rsqrtf(var * (1.0f / (float)HD) + 1e-5f);
  const int h = tid * 4;
  float* y = Y + (size_t)row * HD;
  y[h + 0] = dx * inv * g[h + 0] + bt[h + 0];
  y[h + 1] = dy * inv * g[h + 1] + bt[h + 1];
  y[h + 2] = dz * inv * g[h + 2] + bt[h + 2];
  y[h + 3] = dw * inv * g[h + 3] + bt[h + 3];
}

// pair_feat (192,2048) = [src|dst]; also seed src into src_cat (192,1027)
__global__ __launch_bounds__(256)
void pair_build(const float* __restrict__ gt, float* __restrict__ pair,
                float* __restrict__ srccat) {
  const int r = blockIdx.x, tid = threadIdx.x;
  const int b = r / 3, i = r % 3;
  const float* src = gt + ((size_t)b * 4 + i) * HD;
  const float* dst = src + HD;
  for (int h = tid; h < HD; h += 256) {
    const float sv = src[h];
    pair[(size_t)r * 2048 + h] = sv;
    pair[(size_t)r * 2048 + 1024 + h] = dst[h];
    srccat[(size_t)r * 1027 + h] = sv;
  }
}

// tanh of rh2 output -> relative_position output + src_cat tail columns
__global__ __launch_bounds__(256)
void relpos_finish(const float* __restrict__ rp_raw, float* __restrict__ out_rp,
                   float* __restrict__ srccat) {
  const int idx = blockIdx.x * 256 + threadIdx.x;
  if (idx < 576) {
    const float t = tanhf(rp_raw[idx]);
    out_rp[idx] = t;
    const int r = idx / 3, j = idx % 3;
    srccat[(size_t)r * 1027 + 1024 + j] = t;
  }
}

// cat (64,2048) = [ mean over 4 graph_tokens | mean over 3 rel_tokens ]
__global__ __launch_bounds__(256)
void means_build(const float* __restrict__ gt, const float* __restrict__ rt,
                 float* __restrict__ cat) {
  const int b = blockIdx.x, tid = threadIdx.x;
  for (int h = tid; h < HD; h += 256) {
    const float gm = 0.25f * (gt[((size_t)b * 4 + 0) * HD + h] + gt[((size_t)b * 4 + 1) * HD + h] +
                              gt[((size_t)b * 4 + 2) * HD + h] + gt[((size_t)b * 4 + 3) * HD + h]);
    const float rm = (1.0f / 3.0f) * (rt[((size_t)b * 3 + 0) * HD + h] +
                                      rt[((size_t)b * 3 + 1) * HD + h] +
                                      rt[((size_t)b * 3 + 2) * HD + h]);
    cat[(size_t)b * 2048 + h] = gm;
    cat[(size_t)b * 2048 + 1024 + h] = rm;
  }
}

// 16-wide softmax, one thread per row (64 rows)
__global__ __launch_bounds__(64)
void softmax16(const float* __restrict__ logits, float* __restrict__ attn) {
  const int r = blockIdx.x * blockDim.x + threadIdx.x;
  if (r < 64) {
    const float* x = logits + r * 16;
    float m = x[0];
    for (int j = 1; j < 16; ++j) m = fmaxf(m, x[j]);
    float e[16], s = 0.f;
    for (int j = 0; j < 16; ++j) { e[j] = expf(x[j] - m); s += e[j]; }
    const float inv = 1.0f / s;
    for (int j = 0; j < 16; ++j) attn[r * 16 + j] = e[j] * inv;
  }
}

// ---------------------------------------------------------------------------
extern "C" void kernel_launch(void* const* d_in, const int* in_sizes, int n_in,
                              void* d_out, int out_size, void* d_ws, size_t ws_size,
                              hipStream_t stream) {
  (void)in_sizes; (void)n_in; (void)out_size; (void)ws_size;

  const float* spa      = (const float*)d_in[0];
  const int*   mask     = (const int*)d_in[1];
  const float* ps_w     = (const float*)d_in[2];
  const float* ps_b     = (const float*)d_in[3];
  const float* pn_g     = (const float*)d_in[4];
  const float* pn_b     = (const float*)d_in[5];
  const float* inproj_w = (const float*)d_in[6];
  const float* inproj_b = (const float*)d_in[7];
  const float* outprj_w = (const float*)d_in[8];
  const float* outprj_b = (const float*)d_in[9];
  const float* gn_g     = (const float*)d_in[10];
  const float* gn_b     = (const float*)d_in[11];
  const float* rh1_w    = (const float*)d_in[12];
  const float* rh1_b    = (const float*)d_in[13];
  const float* rh2_w    = (const float*)d_in[14];
  const float* rh2_b    = (const float*)d_in[15];
  const float* rtp_w    = (const float*)d_in[16];
  const float* rtp_b    = (const float*)d_in[17];
  const float* rtp_g    = (const float*)d_in[18];
  const float* rtp_b2   = (const float*)d_in[19];
  const float* of_w     = (const float*)d_in[20];
  const float* of_b     = (const float*)d_in[21];
  const float* of_g     = (const float*)d_in[22];
  const float* of_b2    = (const float*)d_in[23];
  const float* tbank    = (const float*)d_in[24];
  const float* tq_w     = (const float*)d_in[25];
  const float* tq_b     = (const float*)d_in[26];
  const float* tq_g     = (const float*)d_in[27];
  const float* tq_b2    = (const float*)d_in[28];
  const float* tlh_w    = (const float*)d_in[29];
  const float* tlh_b    = (const float*)d_in[30];
  const float* tn_g     = (const float*)d_in[31];
  const float* tn_b     = (const float*)d_in[32];

  float* out = (float*)d_out;
  float* o_pt   = out;             // (64,4,1024)
  float* o_gt   = out + 262144;    // (64,4,1024)
  float* o_rp   = out + 524288;    // (64,3,3)
  float* o_rel  = out + 524864;    // (64,3,1024)
  float* o_rv   = out + 721472;    // (64,1024)
  float* o_tlog = out + 787008;    // (64,16)
  float* o_tatt = out + 788032;    // (64,16)
  float* o_tvec = out + 789056;    // (64,1024)

  // Workspace regions (floats), overlapped by lifetime: total 1310720 f ~ 5.2MB
  float* ws = (float*)d_ws;
  float* Q    = ws;            // 786432 floats
  float* Creg = ws + 786432;   // 262144 floats
  float* G    = ws + 1048576;  // 262144 floats

#define GEMM(A_, lda_, Bw_, ldb_, tB_, bias_, C_, ldc_, M_, N_, K4_, Kt_, gel_)       \
  do {                                                                                \
    int tiles_ = (((M_) + 15) / 16) * (((N_) + 15) / 16);                             \
    int blocks_ = (tiles_ + 7) / 8;                                                   \
    gemm_wmma<<<blocks_, 256, 0, stream>>>((A_), (lda_), (Bw_), (ldb_), (tB_),        \
                                           (bias_), (C_), (ldc_), (M_), (N_),         \
                                           (K4_), (Kt_), (gel_));                     \
  } while (0)

  // 1) Fused pointer stage -> pointer_tokens (output 0)
  pointer_stage<<<64, 256, 0, stream>>>(spa, mask, ps_w, ps_b, pn_g, pn_b, o_pt);

  // 2) qkv = pointer_tokens @ inproj_w.T + b   (256 x 3072, K=1024)
  GEMM(o_pt, 1024, inproj_w, 1024, 1, inproj_b, Q, 3072, 256, 3072, 1024, 0, 0);

  // 3) tiny 4-token attention (TDM-staged qkv tile) -> ctx
  attn_stage<<<64, 256, 0, stream>>>(Q, Creg);

  // 4) outproj (256 x 1024, K=1024); residual + LN -> graph_tokens (output 1)
  GEMM(Creg, 1024, outprj_w, 1024, 1, outprj_b, G, 1024, 256, 1024, 1024, 0, 0);
  ln1024<<<256, 256, 0, stream>>>(G, o_pt, gn_g, gn_b, o_gt);

  // 5) pair features + relative-position head
  float* pair = Q;                 // (192,2048)
  float* scat = Q + 393216;        // (192,1027)
  pair_build<<<192, 256, 0, stream>>>(o_gt, pair, scat);
  float* h1 = Creg;                // (192,1024), gelu fused
  GEMM(pair, 2048, rh1_w, 2048, 1, rh1_b, h1, 1024, 192, 1024, 2048, 0, 1);
  float* rpraw = G;                // (192,3)
  GEMM(h1, 1024, rh2_w, 1024, 1, rh2_b, rpraw, 3, 192, 3, 1024, 0, 0);
  relpos_finish<<<3, 256, 0, stream>>>(rpraw, o_rp, scat);  // output 2

  // 6) rel_tokens = LN(gelu(src_cat @ rtp_w.T + b))  (K = 1024 + tail 3)
  float* rtpw = Creg;              // overwrite h1
  GEMM(scat, 1027, rtp_w, 1027, 1, rtp_b, rtpw, 1024, 192, 1024, 1024, 3, 1);
  ln1024<<<192, 256, 0, stream>>>(rtpw, nullptr, rtp_g, rtp_b2, o_rel);  // output 3

  // 7) means concat [graph_mean | rel_mean]  (64,2048)
  float* cat = Q;
  means_build<<<64, 256, 0, stream>>>(o_gt, o_rel, cat);

  // 8) template query path
  float* tqraw = Creg;             // (64,1024) gelu fused
  GEMM(cat + 1024, 2048, tq_w, 1024, 1, tq_b, tqraw, 1024, 64, 1024, 1024, 0, 1);
  float* tquery = Creg + 65536;
  ln1024<<<64, 256, 0, stream>>>(tqraw, nullptr, tq_g, tq_b2, tquery);
  GEMM(tquery, 1024, tlh_w, 1024, 1, tlh_b, o_tlog, 16, 64, 16, 1024, 0, 0);  // output 5
  softmax16<<<1, 64, 0, stream>>>(o_tlog, o_tatt);                            // output 6
  float* tvraw = Creg + 131072;
  GEMM(o_tatt, 16, tbank, 1024, 0, nullptr, tvraw, 1024, 64, 1024, 16, 0, 0);
  ln1024<<<64, 256, 0, stream>>>(tvraw, nullptr, tn_g, tn_b, o_tvec);         // output 7

  // 9) relation_vec = LN( LN(gelu(cat @ of_w.T + b)) + template_vec )
  float* ofraw = Creg + 196608;    // (64,1024) gelu fused
  GEMM(cat, 2048, of_w, 2048, 1, of_b, ofraw, 1024, 64, 1024, 2048, 0, 1);
  float* rv1 = G;
  ln1024<<<64, 256, 0, stream>>>(ofraw, nullptr, of_g, of_b2, rv1);
  ln1024<<<64, 256, 0, stream>>>(rv1, o_tvec, tn_g, tn_b, o_rv);              // output 4

#undef GEMM
}
